// MyCNNSelfAttentionBaseModule_41154376631134
// MI455X (gfx1250) — matrix operand order repaired and checked by reading, not compiled, not run on hardware
//
#include <hip/hip_runtime.h>
#include <hip/hip_bf16.h>

typedef __attribute__((ext_vector_type(16))) __bf16 v16bf;
typedef __attribute__((ext_vector_type(8)))  float  v8f;
typedef __attribute__((ext_vector_type(4)))  float  v4f;

// Problem constants: B=16,H=64,W=64,C=256,F=256,A=32 ; M = B*H*W = 65536
#define NPIX   65536
#define CCH    256
#define FCH    256
#define ACH    32
#define BN_EPS 1e-3f

// ---------------- workspace layout (bytes) ----------------
#define WS_W3F   0
#define WS_WA1F  131072
#define WS_WA2F  147456
#define WS_BN1S  163840
#define WS_BN1B  164864
#define WS_BN2S  165888
#define WS_BN2B  166016
#define WS_X3    196608
#define WS_AW    (196608ull + 67108864ull)

// ---------------------------------------------------------------------------
// Kernel 0: repack weights into WMMA B-fragment order (bf16) + fold BN params.
// B 32x16 bf16 layout: lane<16 -> col = 16*nt + lane, K = 32*ks + {0..7, 16..23}
//                      lane>=16 -> col = 16*nt + lane-16, K = 32*ks + {8..15, 24..31}
// ---------------------------------------------------------------------------
__global__ __launch_bounds__(256) void prep_kernel(
    const float* __restrict__ w3,  const float* __restrict__ wa1, const float* __restrict__ wa2,
    const float* __restrict__ g1,  const float* __restrict__ be1,
    const float* __restrict__ mu1, const float* __restrict__ va1,
    const float* __restrict__ g2,  const float* __restrict__ be2,
    const float* __restrict__ mu2, const float* __restrict__ va2,
    __bf16* __restrict__ w3f, __bf16* __restrict__ wa1f, __bf16* __restrict__ wa2f,
    float* __restrict__ bn1s, float* __restrict__ bn1b,
    float* __restrict__ bn2s, float* __restrict__ bn2b)
{
    const int tid = blockIdx.x * 256 + threadIdx.x;
    if (tid < 4096) {                               // w3: 16 ntiles x 8 ksteps x 32 lanes
        const int lane = tid & 31, ks = (tid >> 5) & 7, nt = tid >> 8;
        const int col = nt * 16 + (lane & 15);
        const int kb  = ks * 32 + ((lane < 16) ? 0 : 8);
        #pragma unroll
        for (int e = 0; e < 16; ++e)
            w3f[tid * 16 + e] = (__bf16)w3[(kb + e + (e >= 8 ? 8 : 0)) * FCH + col];
    } else if (tid < 4608) {                        // w_a1: 2 x 8 x 32
        const int f = tid - 4096;
        const int lane = f & 31, ks = (f >> 5) & 7, nt = f >> 8;
        const int col = nt * 16 + (lane & 15);
        const int kb  = ks * 32 + ((lane < 16) ? 0 : 8);
        #pragma unroll
        for (int e = 0; e < 16; ++e)
            wa1f[f * 16 + e] = (__bf16)wa1[(kb + e + (e >= 8 ? 8 : 0)) * ACH + col];
    } else if (tid < 5120) {                        // w_a2: 16 ntiles x 1 kstep x 32
        const int f = tid - 4608;
        const int lane = f & 31, nt = f >> 5;
        const int col = nt * 16 + (lane & 15);
        const int kb  = (lane < 16) ? 0 : 8;
        #pragma unroll
        for (int e = 0; e < 16; ++e)
            wa2f[f * 16 + e] = (__bf16)wa2[(kb + e + (e >= 8 ? 8 : 0)) * FCH + col];
    } else if (tid < 5376) {                        // BN1 fold: 256 channels
        const int c = tid - 5120;
        const float s = g1[c] * rsqrtf(va1[c] + BN_EPS);
        bn1s[c] = s; bn1b[c] = be1[c] - mu1[c] * s;
    } else if (tid < 5408) {                        // BN2 fold: 32 channels
        const int c = tid - 5376;
        const float s = g2[c] * rsqrtf(va2[c] + BN_EPS);
        bn2s[c] = s; bn2b[c] = be2[c] - mu2[c] * s;
    }
}

// ---------------------------------------------------------------------------
// Kernel 1: fused value GEMM (x@w3+b3) + attention path. One workgroup
// (8 waves, wave32) per 16 pixel-rows. x-tile staged in LDS once (raw +
// BN1-folded bf16); A-fragments then come from ds_load_b128 pairs.
// ---------------------------------------------------------------------------
__global__ __launch_bounds__(256) void fused_gemm_attn(
    const float* __restrict__ x,   const float* __restrict__ b3,
    const float* __restrict__ ba1, const float* __restrict__ ba2,
    const __bf16* __restrict__ w3f, const __bf16* __restrict__ wa1f,
    const __bf16* __restrict__ wa2f,
    const float* __restrict__ bn1s, const float* __restrict__ bn1b,
    const float* __restrict__ bn2s, const float* __restrict__ bn2b,
    float* __restrict__ x3, float* __restrict__ aw)
{
    __shared__ __bf16 xraw[16 * 256];     // x tile, bf16 (8 KB)
    __shared__ __bf16 xbn [16 * 256];     // BN1-folded x tile, bf16 (8 KB)
    __shared__ __bf16 a1tile[16 * 32];    // bn2(relu(a1)) in bf16 for GEMM2
    __shared__ float  axs[16 * 256];      // pre-softmax logits (16 KB)
    __shared__ float  red[16 * 16];       // per-row reduction scratch

    const int tid   = threadIdx.x;
    const int wave  = __builtin_amdgcn_readfirstlane(tid >> 5);  // scalar wave id
    const int lane  = tid & 31;
    const int lcol  = lane & 15;
    const int lhalf = lane >> 4;          // 0: lanes 0-15, 1: lanes 16-31
    const int row0  = blockIdx.x * 16;

    // -------- stage x tile into LDS (each thread: 16 floats, non-temporal) ----
    {
        const int lr = tid >> 4;           // row 0..15
        const int lc = (tid & 15) * 16;    // col base
        const float* src = x + (size_t)(row0 + lr) * CCH + lc;
        #pragma unroll
        for (int e = 0; e < 16; e += 4) {
            const v4f v = __builtin_nontemporal_load((const v4f*)(src + e));
            const int d = lr * 256 + lc + e;
            #pragma unroll
            for (int j = 0; j < 4; ++j) {
                const float s = bn1s[lc + e + j];
                const float o = bn1b[lc + e + j];
                xraw[d + j] = (__bf16)v[j];
                xbn [d + j] = (__bf16)(v[j] * s + o);
            }
        }
    }
    __syncthreads();

    const v16bf* w3fv  = (const v16bf*)w3f;
    const v16bf* wa1fv = (const v16bf*)wa1f;
    const v16bf* wa2fv = (const v16bf*)wa2f;

    v8f acc[2] = {v8f{}, v8f{}};
    v8f acca   = {};

    const __bf16* arow = &xraw[lcol * 256];   // A-frag row for this lane
    const __bf16* brow = &xbn [lcol * 256];

    for (int ks = 0; ks < 8; ++ks) {
        const int c1 = ks * 32 + lhalf * 8;   // channels for elements 0..7
        const int c2 = c1 + 16;               // channels for elements 8..15

        v16bf av;                              // 2x ds_load_b128
        #pragma unroll
        for (int e = 0; e < 8; ++e) { av[e] = arow[c1 + e]; av[8 + e] = arow[c2 + e]; }

        // value path: two n-tiles per wave
        v16bf bv0 = w3fv[((wave * 2 + 0) * 8 + ks) * 32 + lane];
        v16bf bv1 = w3fv[((wave * 2 + 1) * 8 + ks) * 32 + lane];
        acc[0] = __builtin_amdgcn_wmma_f32_16x16x32_bf16(false, av, false, bv0,
                                                         (short)0, acc[0], false, false);
        acc[1] = __builtin_amdgcn_wmma_f32_16x16x32_bf16(false, av, false, bv1,
                                                         (short)0, acc[1], false, false);

        // attention GEMM1 (BN1-folded A), waves 0-1 only (scalar-uniform branch)
        if (wave < 2) {
            v16bf avbn;
            #pragma unroll
            for (int e = 0; e < 8; ++e) { avbn[e] = brow[c1 + e]; avbn[8 + e] = brow[c2 + e]; }
            v16bf bva = wa1fv[(wave * 8 + ks) * 32 + lane];
            acca = __builtin_amdgcn_wmma_f32_16x16x32_bf16(false, avbn, false, bva,
                                                           (short)0, acca, false, false);
        }
    }

    // x3 = acc + b3  (C/D layout: vgpr r -> row r (lanes<16) / r+8 (lanes>=16))
    #pragma unroll
    for (int t = 0; t < 2; ++t) {
        const int col  = wave * 32 + t * 16 + lcol;
        const float bb = b3[col];
        #pragma unroll
        for (int r = 0; r < 8; ++r) {
            const int row = r + lhalf * 8;
            x3[(size_t)(row0 + row) * FCH + col] = acc[t][r] + bb;
        }
    }

    // a1 -> bias, relu, BN2-fold, bf16, stage in LDS
    if (wave < 2) {
        const int col = wave * 16 + lcol;
        const float bb = ba1[col], s2 = bn2s[col], o2 = bn2b[col];
        #pragma unroll
        for (int r = 0; r < 8; ++r) {
            const int row = r + lhalf * 8;
            float v = acca[r] + bb;
            v = v > 0.f ? v : 0.f;
            a1tile[row * 32 + col] = (__bf16)(v * s2 + o2);
        }
    }
    __syncthreads();

    // GEMM2: K=32, one WMMA per n-tile; each wave does 2 n-tiles
    v16bf afrag;
    {
        const int kb = lhalf * 8;
        #pragma unroll
        for (int e = 0; e < 8; ++e) {
            afrag[e]     = a1tile[lcol * 32 + kb + e];
            afrag[8 + e] = a1tile[lcol * 32 + kb + 16 + e];
        }
    }
    #pragma unroll
    for (int t = 0; t < 2; ++t) {
        const int nt = wave * 2 + t;
        v16bf bv = wa2fv[nt * 32 + lane];
        v8f z = {};
        v8f acc2 = __builtin_amdgcn_wmma_f32_16x16x32_bf16(false, afrag, false, bv,
                                                           (short)0, z, false, false);
        const int col = nt * 16 + lcol;
        const float bb = ba2[col];
        #pragma unroll
        for (int r = 0; r < 8; ++r) {
            const int row = r + lhalf * 8;
            float v = acc2[r] + bb;
            axs[row * 256 + col] = v > 0.f ? v : 0.f;
        }
    }
    __syncthreads();

    // softmax over 256 channels per row: 16 threads per row, 16 cols each
    const int srow = tid >> 4;
    const int chnk = tid & 15;
    float vals[16];
    #pragma unroll
    for (int e = 0; e < 16; ++e) vals[e] = axs[srow * 256 + chnk * 16 + e];
    float pm = vals[0];
    #pragma unroll
    for (int e = 1; e < 16; ++e) pm = fmaxf(pm, vals[e]);
    red[srow * 16 + chnk] = pm;
    __syncthreads();
    float rowm = red[srow * 16];
    #pragma unroll
    for (int e = 1; e < 16; ++e) rowm = fmaxf(rowm, red[srow * 16 + e]);
    __syncthreads();
    float ps = 0.f;
    #pragma unroll
    for (int e = 0; e < 16; ++e) { vals[e] = __expf(vals[e] - rowm); ps += vals[e]; }
    red[srow * 16 + chnk] = ps;
    __syncthreads();
    float rows = 0.f;
    #pragma unroll
    for (int e = 0; e < 16; ++e) rows += red[srow * 16 + e];
    const float inv = 1.0f / rows;
    #pragma unroll
    for (int e = 0; e < 16; ++e)
        aw[(size_t)(row0 + srow) * FCH + chnk * 16 + e] = vals[e] * inv;
}

// ---------------------------------------------------------------------------
// Kernel 2: out = aw * boxsum3x3(x3). x3/aw are L2-resident (192MB L2) so the
// 9x reads are cache hits; out is written once -> non-temporal store.
// ---------------------------------------------------------------------------
__global__ __launch_bounds__(256) void finalize_kernel(
    const float* __restrict__ x3, const float* __restrict__ aw,
    float* __restrict__ out)
{
    const int idx = blockIdx.x * 256 + threadIdx.x;   // NPIX * FCH / 4 threads
    const int p  = idx >> 6;
    const int c0 = (idx & 63) << 2;
    const int b  = p >> 12;
    const int h  = (p >> 6) & 63;
    const int w  = p & 63;
    v4f s = {0.f, 0.f, 0.f, 0.f};
    #pragma unroll
    for (int dh = -1; dh <= 1; ++dh) {
        const int hh = h + dh;
        if (hh < 0 || hh > 63) continue;
        #pragma unroll
        for (int dw = -1; dw <= 1; ++dw) {
            const int ww = w + dw;
            if (ww < 0 || ww > 63) continue;
            const size_t q = (size_t)(((b << 12) | (hh << 6) | ww)) << 8;
            const v4f v = *(const v4f*)(x3 + q + c0);
            s += v;
        }
    }
    const size_t pq = ((size_t)p << 8) + c0;
    const v4f a = *(const v4f*)(aw + pq);
    const v4f o = a * s;
    __builtin_nontemporal_store(o, (v4f*)(out + pq));
}

// ---------------------------------------------------------------------------
extern "C" void kernel_launch(void* const* d_in, const int* in_sizes, int n_in,
                              void* d_out, int out_size, void* d_ws, size_t ws_size,
                              hipStream_t stream)
{
    const float* x   = (const float*)d_in[0];
    const float* w3  = (const float*)d_in[1];
    const float* b3  = (const float*)d_in[2];
    const float* g1  = (const float*)d_in[3];
    const float* be1 = (const float*)d_in[4];
    const float* mu1 = (const float*)d_in[5];
    const float* va1 = (const float*)d_in[6];
    const float* wa1 = (const float*)d_in[7];
    const float* ba1 = (const float*)d_in[8];
    const float* g2  = (const float*)d_in[9];
    const float* be2 = (const float*)d_in[10];
    const float* mu2 = (const float*)d_in[11];
    const float* va2 = (const float*)d_in[12];
    const float* wa2 = (const float*)d_in[13];
    const float* ba2 = (const float*)d_in[14];

    char* ws = (char*)d_ws;
    __bf16* w3f  = (__bf16*)(ws + WS_W3F);
    __bf16* wa1f = (__bf16*)(ws + WS_WA1F);
    __bf16* wa2f = (__bf16*)(ws + WS_WA2F);
    float* bn1s  = (float*)(ws + WS_BN1S);
    float* bn1b  = (float*)(ws + WS_BN1B);
    float* bn2s  = (float*)(ws + WS_BN2S);
    float* bn2b  = (float*)(ws + WS_BN2B);
    float* x3    = (float*)(ws + WS_X3);
    float* aw    = (float*)(ws + WS_AW);

    prep_kernel<<<22, 256, 0, stream>>>(w3, wa1, wa2, g1, be1, mu1, va1,
                                        g2, be2, mu2, va2,
                                        w3f, wa1f, wa2f, bn1s, bn1b, bn2s, bn2b);

    fused_gemm_attn<<<NPIX / 16, 256, 0, stream>>>(x, b3, ba1, ba2,
                                                   w3f, wa1f, wa2f,
                                                   bn1s, bn1b, bn2s, bn2b,
                                                   x3, aw);

    finalize_kernel<<<(NPIX * FCH / 4) / 256, 256, 0, stream>>>(x3, aw,
                                                                (float*)d_out);
}
